// DisentangledMultiHeadAttention_65309272703589
// MI455X (gfx1250) — matrix operand order repaired and checked by reading, not compiled, not run on hardware
//
#include <hip/hip_runtime.h>

// ---------------------------------------------------------------------------
// DisentangledMultiHeadAttention forward for MI455X (gfx1250, wave32, WMMA)
//   B=8, T=2048, C=4096, H=2, D=2048.  Outputs: y (B,T,C) fp32 ++ atten
//   (B,H,T,T) fp32.  All GEMMs run through v_wmma_f32_16x16x32_bf16 with
//   fp32 accumulation.  Double-buffered LDS staging; bf16-source tiles use
//   GLOBAL_LOAD_ASYNC_TO_LDS (ASYNCcnt) when the toolchain exposes it.
// ---------------------------------------------------------------------------

typedef __bf16 v16bf __attribute__((ext_vector_type(16)));
typedef __bf16 v8bf  __attribute__((ext_vector_type(8)));
typedef float  v8f   __attribute__((ext_vector_type(8)));
typedef int    v4i   __attribute__((vector_size(4 * sizeof(int))));
typedef __attribute__((address_space(1))) v4i* gv4i_p;
typedef __attribute__((address_space(3))) v4i* lv4i_p;

#if __has_builtin(__builtin_amdgcn_global_load_async_to_lds_b128)
#define HAVE_ASYNC_LDS 1
#if __has_builtin(__builtin_amdgcn_s_wait_asynccnt)
#define WAIT_ASYNC() __builtin_amdgcn_s_wait_asynccnt(0)
#else
#define WAIT_ASYNC() asm volatile("s_wait_asynccnt 0x0" ::: "memory")
#endif
#else
#define HAVE_ASYNC_LDS 0
#define WAIT_ASYNC()
#endif

__device__ __forceinline__ unsigned short f32_to_bf16(float f) {
    unsigned int u = __float_as_uint(f);
    u += 0x7FFFu + ((u >> 16) & 1u);      // round-to-nearest-even
    return (unsigned short)(u >> 16);
}

#define BM 128
#define BN 128
#define BK 32
#define LDSS 40   // LDS row stride in halfwords (32 used + 8 pad -> no bank conflicts)
#define BUFSZ (BM * LDSS)

// Generic tiled WMMA GEMM:  Cz = scale * Az @ Bz^T(+bias)   (NT path)
//                           Cz = scale * Az @ Bz            (B_KXN path, B staged transposed)
// A: M x K row-major (fp32 or bf16).  B: N x K row-major (NT) or K x N (B_KXN, bf16 only).
// Per-z offsets (elements): off = (z/Hdiv)*T1 + (z%Hdiv)*T2.
template<bool A_BF16, bool B_BF16, bool B_KXN, bool C_BF16, bool HAS_BIAS>
__global__ __launch_bounds__(256)
void gemm_wmma(const void* __restrict__ A, const void* __restrict__ B,
               void* __restrict__ Cp, const float* __restrict__ bias,
               int K, int lda, int ldb, int ldc, float scale, int Hdiv,
               long long aT1, long long aT2, long long bT1, long long bT2,
               long long cT1, long long cT2, long long biasT2)
{
    constexpr bool A_ASYNC = (HAVE_ASYNC_LDS != 0) && A_BF16;
    constexpr bool B_ASYNC = (HAVE_ASYNC_LDS != 0) && B_BF16 && !B_KXN;
    constexpr bool ANY_ASYNC = A_ASYNC || B_ASYNC;

    __shared__ alignas(16) unsigned short sA[2 * BUFSZ];
    __shared__ alignas(16) unsigned short sB[2 * BUFSZ];

    const int z  = blockIdx.z;
    const int bz = z / Hdiv;
    const int hz = z - bz * Hdiv;
    const long long aOff = (long long)bz * aT1 + (long long)hz * aT2;
    const long long bOff = (long long)bz * bT1 + (long long)hz * bT2;
    const long long cOff = (long long)bz * cT1 + (long long)hz * cT2;

    const float*          Af  = (const float*)A          + aOff;
    const unsigned short* Ah  = (const unsigned short*)A + aOff;
    const float*          Bf  = (const float*)B          + bOff;
    const unsigned short* Bh  = (const unsigned short*)B + bOff;
    float*                Cf  = (float*)Cp               + cOff;
    unsigned short*       Ch  = (unsigned short*)Cp      + cOff;
    const float* biasP = HAS_BIAS ? (bias + (long long)hz * biasT2) : nullptr;

    const int blockRow = blockIdx.y * BM;
    const int blockCol = blockIdx.x * BN;

    const int tid    = threadIdx.x;
    const int lane   = tid & 31;
    const int wid    = tid >> 5;
    const int waveM  = wid >> 2;        // 0..1  -> 64 rows each
    const int waveN  = wid & 3;         // 0..3  -> 32 cols each
    const int khalf  = lane >> 4;       // lane half selects K sub-block
    const int l16    = lane & 15;

    v8f acc[4][2];
#pragma unroll
    for (int mi = 0; mi < 4; ++mi)
#pragma unroll
        for (int ni = 0; ni < 2; ++ni)
#pragma unroll
            for (int e = 0; e < 8; ++e) acc[mi][ni][e] = 0.0f;

    // staging registers (only the arrays used by this instantiation survive)
    float4 aRegF[4]; uint2 aRegH[4];
    float4 bRegF[4]; uint2 bRegH[4];

    // ---- issue: start global fetches for K-slice kt into buffer bufi ----
    auto issueA = [&](int kt, int bufi) {
        const int k0 = kt * BK;
        if constexpr (A_ASYNC) {
#if HAVE_ASYNC_LDS
            unsigned short* sAb = sA + bufi * BUFSZ;
#pragma unroll
            for (int i = 0; i < 2; ++i) {
                const int id = tid + i * 256;       // 0..511 b128 chunks
                const int r  = id >> 2;             // 0..127
                const int c  = (id & 3) << 3;       // 0,8,16,24 halfwords
                const long long gi = (long long)(blockRow + r) * lda + (k0 + c);
                __builtin_amdgcn_global_load_async_to_lds_b128(
                    (gv4i_p)(Ah + gi), (lv4i_p)&sAb[r * LDSS + c], 0, 0);
            }
#endif
        } else {
#pragma unroll
            for (int i = 0; i < 4; ++i) {
                const int id = tid + i * 256;
                const int r  = id >> 3;
                const int c  = (id & 7) << 2;
                const long long gi = (long long)(blockRow + r) * lda + (k0 + c);
                if constexpr (A_BF16) aRegH[i] = *(const uint2*)(Ah + gi);
                else                  aRegF[i] = *(const float4*)(Af + gi);
            }
        }
    };
    auto issueB = [&](int kt, int bufi) {
        const int k0 = kt * BK;
        if constexpr (B_ASYNC) {
#if HAVE_ASYNC_LDS
            unsigned short* sBb = sB + bufi * BUFSZ;
#pragma unroll
            for (int i = 0; i < 2; ++i) {
                const int id = tid + i * 256;
                const int r  = id >> 2;
                const int c  = (id & 3) << 3;
                const long long gi = (long long)(blockCol + r) * ldb + (k0 + c);
                __builtin_amdgcn_global_load_async_to_lds_b128(
                    (gv4i_p)(Bh + gi), (lv4i_p)&sBb[r * LDSS + c], 0, 0);
            }
#endif
        } else if constexpr (B_KXN) {               // bf16 K x N, raw fetch
#pragma unroll
            for (int i = 0; i < 4; ++i) {
                const int id = tid + i * 256;
                const int kk = id >> 5;             // 0..31
                const int n  = (id & 31) << 2;      // 0..124
                const long long gi = (long long)(k0 + kk) * ldb + (blockCol + n);
                bRegH[i] = *(const uint2*)(Bh + gi);
            }
        } else {
#pragma unroll
            for (int i = 0; i < 4; ++i) {
                const int id = tid + i * 256;
                const int r  = id >> 3;
                const int c  = (id & 7) << 2;
                const long long gi = (long long)(blockCol + r) * ldb + (k0 + c);
                if constexpr (B_BF16) bRegH[i] = *(const uint2*)(Bh + gi);
                else                  bRegF[i] = *(const float4*)(Bf + gi);
            }
        }
    };

    // ---- commit: convert/scatter staged registers into LDS buffer bufi ----
    auto commitA = [&](int bufi) {
        if constexpr (!A_ASYNC) {
            unsigned short* sAb = sA + bufi * BUFSZ;
#pragma unroll
            for (int i = 0; i < 4; ++i) {
                const int id = tid + i * 256;
                const int r  = id >> 3;
                const int c  = (id & 7) << 2;
                uint2 packed;
                if constexpr (A_BF16) packed = aRegH[i];
                else {
                    packed.x = (unsigned)f32_to_bf16(aRegF[i].x) | ((unsigned)f32_to_bf16(aRegF[i].y) << 16);
                    packed.y = (unsigned)f32_to_bf16(aRegF[i].z) | ((unsigned)f32_to_bf16(aRegF[i].w) << 16);
                }
                *(uint2*)&sAb[r * LDSS + c] = packed;
            }
        }
    };
    auto commitB = [&](int bufi) {
        if constexpr (B_ASYNC) {
            // nothing: data lands in LDS via async DMA
        } else if constexpr (B_KXN) {
            unsigned short* sBb = sB + bufi * BUFSZ;
#pragma unroll
            for (int i = 0; i < 4; ++i) {
                const int id = tid + i * 256;
                const int kk = id >> 5;
                const int n  = (id & 31) << 2;
                sBb[(n + 0) * LDSS + kk] = (unsigned short)(bRegH[i].x);
                sBb[(n + 1) * LDSS + kk] = (unsigned short)(bRegH[i].x >> 16);
                sBb[(n + 2) * LDSS + kk] = (unsigned short)(bRegH[i].y);
                sBb[(n + 3) * LDSS + kk] = (unsigned short)(bRegH[i].y >> 16);
            }
        } else {
            unsigned short* sBb = sB + bufi * BUFSZ;
#pragma unroll
            for (int i = 0; i < 4; ++i) {
                const int id = tid + i * 256;
                const int r  = id >> 3;
                const int c  = (id & 7) << 2;
                uint2 packed;
                if constexpr (B_BF16) packed = bRegH[i];
                else {
                    packed.x = (unsigned)f32_to_bf16(bRegF[i].x) | ((unsigned)f32_to_bf16(bRegF[i].y) << 16);
                    packed.y = (unsigned)f32_to_bf16(bRegF[i].z) | ((unsigned)f32_to_bf16(bRegF[i].w) << 16);
                }
                *(uint2*)&sBb[r * LDSS + c] = packed;
            }
        }
    };

    // ---- compute: fragments per ISA VGPR layouts + 8 WMMAs ----
    auto compute = [&](int bufi) {
        const unsigned short* sAb = sA + bufi * BUFSZ;
        const unsigned short* sBb = sB + bufi * BUFSZ;
        v16bf afrag[4], bfrag[2];
#pragma unroll
        for (int mi = 0; mi < 4; ++mi) {
            const int m = waveM * 64 + mi * 16 + l16;
            const unsigned short* pa = &sAb[m * LDSS + 8 * khalf];
            const v8bf lo = *(const v8bf*)(pa);
            const v8bf hi = *(const v8bf*)(pa + 16);
#pragma unroll
            for (int e = 0; e < 8; ++e) { afrag[mi][e] = lo[e]; afrag[mi][e + 8] = hi[e]; }
        }
#pragma unroll
        for (int ni = 0; ni < 2; ++ni) {
            const int n = waveN * 32 + ni * 16 + l16;
            const unsigned short* pb = &sBb[n * LDSS + 8 * khalf];
            const v8bf lo = *(const v8bf*)(pb);
            const v8bf hi = *(const v8bf*)(pb + 16);
#pragma unroll
            for (int e = 0; e < 8; ++e) { bfrag[ni][e] = lo[e]; bfrag[ni][e + 8] = hi[e]; }
        }
#pragma unroll
        for (int mi = 0; mi < 4; ++mi)
#pragma unroll
            for (int ni = 0; ni < 2; ++ni)
                acc[mi][ni] = __builtin_amdgcn_wmma_f32_16x16x32_bf16(
                    false, afrag[mi], false, bfrag[ni], (short)0, acc[mi][ni], false, false);
    };

    const int ktiles = K / BK;

    // prologue: stage K-slice 0 into buffer 0
    issueA(0, 0); issueB(0, 0);
    commitA(0);   commitB(0);
    if constexpr (ANY_ASYNC) WAIT_ASYNC();
    __syncthreads();

    for (int kt = 0; kt < ktiles; ++kt) {
        const int buf  = kt & 1;
        const int nbuf = buf ^ 1;
        const bool more = (kt + 1) < ktiles;
        if (more) { issueA(kt + 1, nbuf); issueB(kt + 1, nbuf); }  // overlap with WMMA
        compute(buf);
        if (more) { commitA(nbuf); commitB(nbuf); }                // waits land here
        if constexpr (ANY_ASYNC) WAIT_ASYNC();
        __syncthreads();
    }

    // ---- epilogue: scale, optional per-column bias, store fp32 or bf16 ----
#pragma unroll
    for (int mi = 0; mi < 4; ++mi) {
#pragma unroll
        for (int ni = 0; ni < 2; ++ni) {
            const int col = blockCol + waveN * 32 + ni * 16 + l16;
            float bv = 0.0f;
            if constexpr (HAS_BIAS) bv = biasP[col];
#pragma unroll
            for (int r = 0; r < 8; ++r) {
                const int row = blockRow + waveM * 64 + mi * 16 + 8 * khalf + r;
                const float v = acc[mi][ni][r] * scale + bv;
                if constexpr (C_BF16)
                    Ch[(long long)row * ldc + col] = f32_to_bf16(v);
                else
                    Cf[(long long)row * ldc + col] = v;
            }
        }
    }
}

// Row softmax over 2048 columns, one 256-thread block per row, in place.
__global__ __launch_bounds__(256)
void softmax_rows(float* __restrict__ p)
{
    float* r = p + (long long)blockIdx.x * 2048;
    const int tid  = threadIdx.x;
    const int lane = tid & 31;
    const int wid  = tid >> 5;
    __shared__ float smax[8], ssum[8];

    float v[8];
    float m = -3.402823466e38f;
#pragma unroll
    for (int i = 0; i < 8; ++i) { v[i] = r[tid + i * 256]; m = fmaxf(m, v[i]); }
#pragma unroll
    for (int off = 16; off > 0; off >>= 1) m = fmaxf(m, __shfl_xor(m, off, 32));
    if (lane == 0) smax[wid] = m;
    __syncthreads();
#pragma unroll
    for (int i = 0; i < 8; ++i) m = fmaxf(m, smax[i]);

    float s = 0.0f;
#pragma unroll
    for (int i = 0; i < 8; ++i) { v[i] = __expf(v[i] - m); s += v[i]; }
#pragma unroll
    for (int off = 16; off > 0; off >>= 1) s += __shfl_xor(s, off, 32);
    if (lane == 0) ssum[wid] = s;
    __syncthreads();
    s = 0.0f;
#pragma unroll
    for (int i = 0; i < 8; ++i) s += ssum[i];

    const float inv = 1.0f / s;
#pragma unroll
    for (int i = 0; i < 8; ++i) r[tid + i * 256] = v[i] * inv;
}

extern "C" void kernel_launch(void* const* d_in, const int* in_sizes, int n_in,
                              void* d_out, int out_size, void* d_ws, size_t ws_size,
                              hipStream_t stream) {
    (void)in_sizes; (void)n_in; (void)out_size; (void)ws_size;

    constexpr int  Bb = 8, T = 2048, Cn = 4096, H = 2, D = 2048;
    constexpr long long BTC = (long long)Bb * T * Cn;      // 67,108,864 elements

    const float* x  = (const float*)d_in[0];
    const float* Wq = (const float*)d_in[1];
    const float* Wk = (const float*)d_in[2];
    const float* Wv = (const float*)d_in[3];
    const float* Wo = (const float*)d_in[4];
    const float* pb = (const float*)d_in[5];

    float* y     = (float*)d_out;          // (B,T,C)
    float* atten = y + BTC;                // (B,H,T,T)

    // workspace (bf16): q | k | v | yh  -> 4 * BTC * 2 bytes = 512 MiB
    unsigned short* q  = (unsigned short*)d_ws;
    unsigned short* k  = q + BTC;
    unsigned short* v  = k + BTC;
    unsigned short* yh = v + BTC;

    const dim3 blk(256);
    const dim3 gproj(Cn / BN, (Bb * T) / BM, 1);           // (32,128,1)
    const dim3 gattn(T / BN, T / BM, Bb * H);              // (16,16,16)
    const float sc = 1.0f / sqrtf((float)D);

    // 1-3) Q,K,V projections: bf16 <- fp32 x @ W^T
    gemm_wmma<false,false,false,true,false><<<gproj, blk, 0, stream>>>(
        x, Wq, q, nullptr, Cn, Cn, Cn, Cn, 1.0f, 1, 0,0, 0,0, 0,0, 0);
    gemm_wmma<false,false,false,true,false><<<gproj, blk, 0, stream>>>(
        x, Wk, k, nullptr, Cn, Cn, Cn, Cn, 1.0f, 1, 0,0, 0,0, 0,0, 0);
    gemm_wmma<false,false,false,true,false><<<gproj, blk, 0, stream>>>(
        x, Wv, v, nullptr, Cn, Cn, Cn, Cn, 1.0f, 1, 0,0, 0,0, 0,0, 0);

    // 4) scores = sc * Q @ K^T + position_bias[h, key]   -> atten (fp32)
    gemm_wmma<true,true,false,false,true><<<gattn, blk, 0, stream>>>(
        q, k, atten, pb, D, Cn, Cn, T, sc, H,
        (long long)T * Cn, D,                 // A: per-b T*C, per-h D
        (long long)T * Cn, D,                 // B: same head striding
        (long long)H * T * T, (long long)T * T,
        D);                                   // bias: per-h D

    // 5) softmax rows (B*H*T rows of T)
    softmax_rows<<<Bb * H * T, blk, 0, stream>>>(atten);

    // 6) yh = atten @ V   (A fp32 converted, B bf16 K x N transposed in LDS)
    gemm_wmma<false,true,true,true,false><<<gattn, blk, 0, stream>>>(
        atten, v, yh, nullptr, T, T, Cn, Cn, 1.0f, H,
        (long long)H * T * T, (long long)T * T,
        (long long)T * Cn, D,
        (long long)T * Cn, D, 0);

    // 7) y = yh @ Wo^T   (fp32 out)
    gemm_wmma<true,false,false,false,false><<<gproj, blk, 0, stream>>>(
        yh, Wo, y, nullptr, Cn, Cn, Cn, Cn, 1.0f, 1, 0,0, 0,0, 0,0, 0);
}
